// Simulator_19602230739700
// MI455X (gfx1250) — matrix operand order repaired and checked by reading, not compiled, not run on hardware
//
#include <hip/hip_runtime.h>
#include <hip/hip_bf16.h>
#include <stdint.h>
#include <stddef.h>

#define N_ATOMS 2048
#define KNN     16
#define E_EDGES (N_ATOMS * KNN)   // 32768
#define HDIM    128
#define L_LAYERS 10
#define N_STEPS  10

typedef __attribute__((ext_vector_type(16))) __bf16 bf16x16;
typedef __attribute__((ext_vector_type(8)))  float  f32x8;

union ABu { uint4 u[2]; bf16x16 v; };

// ---------- helpers ----------
__device__ __forceinline__ unsigned short f2bf(float f) {
  unsigned u = __float_as_uint(f);
  u += 0x7FFFu + ((u >> 16) & 1u);          // round-to-nearest-even
  return (unsigned short)(u >> 16);
}
__device__ __forceinline__ float bf2f(unsigned short h) {
  return __uint_as_float(((unsigned)h) << 16);
}
__device__ __forceinline__ unsigned pcg_hash(unsigned x) {
  x = x * 747796405u + 2891336453u;
  unsigned w = ((x >> ((x >> 28u) + 4u)) ^ x) * 277803737u;
  return (w >> 22u) ^ w;
}

// Workgroup barrier with LDS-only fence: drains DScnt (so staged LDS tiles are
// visible across waves) but deliberately does NOT drain loadcnt, keeping the
// register-destined global prefetch loads in flight across the barrier.
// (__syncthreads() emits s_wait_loadcnt_dscnt 0x0 which serializes the pipeline.)
__device__ __forceinline__ void wg_sync_lds() {
  asm volatile("s_wait_dscnt 0x0\n\t"
               "s_barrier_signal -1\n\t"
               "s_barrier_wait -1" ::: "memory");
}

// ---------- weight packing: fp32 [K,128] -> bf16 B-fragment order ----------
// layout: [kb][ntile(8)][lane(32)][dword(8)]
// lane l: N = ntile*16 + (l&15); Kbase = kb*32 + (l>>4)*16; dword v holds K pair (Kbase+2v, Kbase+2v+1)
__global__ __launch_bounds__(256)
void pack_w_kernel(const float* __restrict__ W, unsigned* __restrict__ out, int Kvalid) {
  int t = blockIdx.x * 256 + threadIdx.x;   // exact grid: (Kpad/32)*256 threads
  int lane = t & 31;
  int nt   = (t >> 5) & 7;
  int kb   = t >> 8;
  int n = nt * 16 + (lane & 15);
  int kbase = kb * 32 + (lane >> 4) * 16;
  unsigned* o = out + (size_t)t * 8;
#pragma unroll
  for (int v = 0; v < 8; ++v) {
    int k0 = kbase + 2 * v, k1 = k0 + 1;
    unsigned lo = (k0 < Kvalid) ? f2bf(W[(size_t)k0 * HDIM + n]) : 0;
    unsigned hi = (k1 < Kvalid) ? f2bf(W[(size_t)k1 * HDIM + n]) : 0;
    o[v] = lo | (hi << 16);
  }
}

// ---------- init: perturb positions, zero velocities ----------
__global__ __launch_bounds__(256)
void init_kernel(const float* __restrict__ pos_in, float* __restrict__ pos,
                 float* __restrict__ vels) {
  int i = blockIdx.x * 256 + threadIdx.x;
  if (i >= N_ATOMS * 3) return;
  unsigned r1 = pcg_hash(2u * (unsigned)i + 1u);
  unsigned r2 = pcg_hash(2u * (unsigned)i + 2u);
  float u1 = ((float)r1 + 1.0f) * 2.3283064e-10f;  // (0,1]
  float u2 = (float)r2 * 2.3283064e-10f;
  float g = sqrtf(-2.0f * logf(u1)) * cosf(6.2831853f * u2);
  pos[i]  = pos_in[i] + 0.1f * g;
  vels[i] = 0.0f;
}

// ---------- kNN: per-thread top-16 insertion ----------
__global__ __launch_bounds__(256)
void knn_kernel(const float* __restrict__ pos, int* __restrict__ senders) {
  int i = blockIdx.x * 256 + threadIdx.x;
  if (i >= N_ATOMS) return;
  float px = pos[i * 3], py = pos[i * 3 + 1], pz = pos[i * 3 + 2];
  float bd[KNN];
  int   bi[KNN];
#pragma unroll
  for (int t = 0; t < KNN; ++t) { bd[t] = 3.0e38f; bi[t] = 0; }
  for (int j = 0; j < N_ATOMS; ++j) {
    if (j == i) continue;
    float dx = px - pos[j * 3], dy = py - pos[j * 3 + 1], dz = pz - pos[j * 3 + 2];
    float d2 = dx * dx + dy * dy + dz * dz;
    if (d2 < bd[KNN - 1]) {
      int t = KNN - 1;
      while (t > 0 && bd[t - 1] > d2) { bd[t] = bd[t - 1]; bi[t] = bi[t - 1]; --t; }
      bd[t] = d2; bi[t] = j;
    }
  }
#pragma unroll
  for (int t = 0; t < KNN; ++t) senders[i * KNN + t] = bi[t];
}

// ---------- edge attributes ea0 = [rel(3), dist] in bf16 ----------
__global__ __launch_bounds__(256)
void edge_attr_kernel(const float* __restrict__ pos, const int* __restrict__ senders,
                      unsigned short* __restrict__ ea0) {
  int e = blockIdx.x * 256 + threadIdx.x;
  if (e >= E_EDGES) return;
  int s = senders[e], r = e >> 4;
  float dx = pos[s * 3]     - pos[r * 3];
  float dy = pos[s * 3 + 1] - pos[r * 3 + 1];
  float dz = pos[s * 3 + 2] - pos[r * 3 + 2];
  float dn = sqrtf(dx * dx + dy * dy + dz * dz);
  unsigned* o = (unsigned*)(ea0 + (size_t)e * 4);
  o[0] = (unsigned)f2bf(dx) | ((unsigned)f2bf(dy) << 16);
  o[1] = (unsigned)f2bf(dz) | ((unsigned)f2bf(dn) << 16);
}

// ---------- segment-sum scatter: agg[senders[e]] += e_row ----------
__global__ __launch_bounds__(256)
void scatter_kernel(const unsigned short* __restrict__ e, const int* __restrict__ senders,
                    float* __restrict__ agg) {
  int ed = blockIdx.x * 256 + threadIdx.x;
  if (ed >= E_EDGES) return;
  int s = senders[ed];
  const uint4* row = (const uint4*)(e + (size_t)ed * HDIM);
  float* dst = agg + (size_t)s * HDIM;
#pragma unroll 4
  for (int c4 = 0; c4 < 16; ++c4) {
    uint4 d = row[c4];
    unsigned w[4] = {d.x, d.y, d.z, d.w};
#pragma unroll
    for (int u = 0; u < 4; ++u) {
      atomicAdd(&dst[c4 * 8 + 2 * u],     bf2f((unsigned short)(w[u] & 0xFFFF)));
      atomicAdd(&dst[c4 * 8 + 2 * u + 1], bf2f((unsigned short)(w[u] >> 16)));
    }
  }
}

// ---------- decoder tail [128x3] + integrate ----------
__global__ __launch_bounds__(256)
void decode_integrate_kernel(const unsigned short* __restrict__ xdec,
                             const float* __restrict__ dW2, const float* __restrict__ db2,
                             const float* __restrict__ mass,
                             float* __restrict__ pos, float* __restrict__ vels) {
  int i = blockIdx.x * 256 + threadIdx.x;
  if (i >= N_ATOMS) return;
  float f0 = db2[0], f1 = db2[1], f2 = db2[2];
  const unsigned short* xr = xdec + (size_t)i * HDIM;
#pragma unroll 8
  for (int k = 0; k < HDIM; ++k) {
    float xv = bf2f(xr[k]);
    f0 += xv * dW2[k * 3];
    f1 += xv * dW2[k * 3 + 1];
    f2 += xv * dW2[k * 3 + 2];
  }
  float im = 1.0f / mass[i];
  float v0 = vels[i * 3]     + f0 * im;
  float v1 = vels[i * 3 + 1] + f1 * im;
  float v2 = vels[i * 3 + 2] + f2 * im;
  vels[i * 3] = v0; vels[i * 3 + 1] = v1; vels[i * 3 + 2] = v2;
  pos[i * 3] += v0; pos[i * 3 + 1] += v1; pos[i * 3 + 2] += v2;
}

// ---------- WMMA GEMM: out[M,128] = act( sum_segs A_s @ W_s + bias ) ----------
struct GemmSeg {
  const void*     A;     // activations, row-major [rows, lda]; bf16 or fp32
  const int*      idx;   // gather indices (mode 1)
  const unsigned* Wp;    // packed bf16 weights for this K range
  int K;                 // valid K (padded to 32 internally with zeros)
  int lda;               // in elements
  int dtype;             // 0 = bf16, 1 = fp32
  int mode;              // 0 = direct row, 1 = idx[row] gather, 2 = row>>4 (receivers)
};

#define LDS_STRIDE 40   // bf16 elems per tile row (32 + 8 pad), 80B keeps 16B alignment

// pick the (segment, kb) pair for flattened k-step t
__device__ __forceinline__ GemmSeg pick_seg(const GemmSeg& s0, const GemmSeg& s1,
                                            const GemmSeg& s2, int t, int n0, int n1,
                                            int& kb) {
  if (t < n0) { kb = t; return s0; }
  if (t < n0 + n1) { kb = t - n0; return s1; }
  kb = t - n0 - n1;
  return s2;
}

// compute this thread's two 8-bf16 quads of the 128x32 A tile (issues global loads)
__device__ __forceinline__ void tile_fill_regs(const GemmSeg& seg, int kb, int tid,
                                               int blockRow, uint4& d0, uint4& d1) {
  const bool vec = (seg.dtype == 0) && ((seg.K & 31) == 0);
#pragma unroll
  for (int q = 0; q < 2; ++q) {
    int qid  = tid + q * 256;      // 0..511
    int row  = qid >> 2;
    int quad = qid & 3;
    int grow = blockRow + row;
    int g = (seg.mode == 1) ? seg.idx[grow] : ((seg.mode == 2) ? (grow >> 4) : grow);
    int k0 = kb * 32 + quad * 8;
    uint4 d;
    if (vec) {
      d = *(const uint4*)((const unsigned short*)seg.A + (size_t)g * seg.lda + k0);
    } else {
      union { unsigned short s[8]; uint4 u; } t8;
#pragma unroll
      for (int u = 0; u < 8; ++u) {
        int k = k0 + u;
        float v = 0.0f;
        if (k < seg.K) {
          v = (seg.dtype == 0)
                  ? bf2f(((const unsigned short*)seg.A)[(size_t)g * seg.lda + k])
                  : ((const float*)seg.A)[(size_t)g * seg.lda + k];
        }
        t8.s[u] = f2bf(v);
      }
      d = t8.u;
    }
    if (q == 0) d0 = d; else d1 = d;
  }
}

// issue 8 global_load_b128 for the four 16-col B fragments of k-step t
__device__ __forceinline__ void load_bfrag(ABu bb[4], const GemmSeg& sg, int kb,
                                           int nt0, int lane) {
#pragma unroll
  for (int nf = 0; nf < 4; ++nf) {
    const uint4* wp =
        (const uint4*)sg.Wp + ((size_t)((kb * 8) + nt0 + nf) * 32 + lane) * 2;
    bb[nf].u[0] = wp[0];
    bb[nf].u[1] = wp[1];
  }
}

// 4 ds_load_b128 (A frags) + 8 WMMA against resident B fragments
__device__ __forceinline__ void compute_tile(f32x8 acc[2][4], const ABu bb[4],
                                             const unsigned short* buf, int m0, int lane) {
  ABu a[2];
  const int koff = (lane >> 4) * 16;
#pragma unroll
  for (int mf = 0; mf < 2; ++mf) {
    int r = m0 + mf * 16 + (lane & 15);
    a[mf].u[0] = *(const uint4*)&buf[r * LDS_STRIDE + koff];
    a[mf].u[1] = *(const uint4*)&buf[r * LDS_STRIDE + koff + 8];
  }
#pragma unroll
  for (int nf = 0; nf < 4; ++nf)
#pragma unroll
    for (int mf = 0; mf < 2; ++mf)
      acc[mf][nf] = __builtin_amdgcn_wmma_f32_16x16x32_bf16(
          false, a[mf].v, false, bb[nf].v, (short)0, acc[mf][nf], false, false);
}

__global__ __launch_bounds__(256)
void gemm128_kernel(GemmSeg s0, GemmSeg s1, GemmSeg s2, int nseg,
                    const float* __restrict__ bias, int relu,
                    unsigned short* __restrict__ out) {
  // double-buffered A tiles: 2 x (128 rows x 32 bf16, padded stride)
  __shared__ __attribute__((aligned(16))) unsigned short smem[2][128 * LDS_STRIDE];
  const int tid  = threadIdx.x;
  const int lane = tid & 31;
  const int wv   = tid >> 5;
  const int m0   = (wv & 3) * 32;    // wave tile: 32 rows x 64 cols
  const int nt0  = (wv >> 2) * 4;    // first 16-col tile
  const int blockRow = blockIdx.x * 128;
  const int srow  = tid >> 2;        // this thread's LDS fill slots
  const int squad = tid & 3;

  const int n0 = (s0.K + 31) >> 5;
  const int n1 = (nseg > 1) ? ((s1.K + 31) >> 5) : 0;
  const int n2 = (nseg > 2) ? ((s2.K + 31) >> 5) : 0;
  const int total = n0 + n1 + n2;

  f32x8 zero = {0.f, 0.f, 0.f, 0.f, 0.f, 0.f, 0.f, 0.f};
  f32x8 acc[2][4];
#pragma unroll
  for (int mf = 0; mf < 2; ++mf)
#pragma unroll
    for (int nf = 0; nf < 4; ++nf) acc[mf][nf] = zero;

  ABu b0[4], b1[4];   // double-buffered B fragments (loop-carried across barriers)

  // prologue: stage A tile 0 and B fragments 0
  {
    int kb;
    GemmSeg sg = pick_seg(s0, s1, s2, 0, n0, n1, kb);
    uint4 d0, d1;
    tile_fill_regs(sg, kb, tid, blockRow, d0, d1);
    load_bfrag(b0, sg, kb, nt0, lane);
    *(uint4*)&smem[0][srow * LDS_STRIDE + squad * 8] = d0;
    *(uint4*)&smem[0][(srow + 64) * LDS_STRIDE + squad * 8] = d1;
  }

  for (int t = 0; t < total; t += 2) {
    // ---- even half: compute tile t from b0; prefetch tile t+1 into b1 ----
    {
      bool hn = (t + 1) < total;
      uint4 p0, p1;
      if (hn) {
        int kb;
        GemmSeg sg = pick_seg(s0, s1, s2, t + 1, n0, n1, kb);
        tile_fill_regs(sg, kb, tid, blockRow, p0, p1);  // A prefetch first...
        load_bfrag(b1, sg, kb, nt0, lane);              // ...so B stays outstanding longer
      }
      wg_sync_lds();
      compute_tile(acc, b0, smem[0], m0, lane);
      if (hn) {
        *(uint4*)&smem[1][srow * LDS_STRIDE + squad * 8] = p0;
        *(uint4*)&smem[1][(srow + 64) * LDS_STRIDE + squad * 8] = p1;
      }
    }
    // ---- odd half: compute tile t+1 from b1; prefetch tile t+2 into b0 ----
    if ((t + 1) < total) {
      bool hn = (t + 2) < total;
      uint4 p0, p1;
      if (hn) {
        int kb;
        GemmSeg sg = pick_seg(s0, s1, s2, t + 2, n0, n1, kb);
        tile_fill_regs(sg, kb, tid, blockRow, p0, p1);
        load_bfrag(b0, sg, kb, nt0, lane);
      }
      wg_sync_lds();
      compute_tile(acc, b1, smem[1], m0, lane);
      if (hn) {
        *(uint4*)&smem[0][srow * LDS_STRIDE + squad * 8] = p0;
        *(uint4*)&smem[0][(srow + 64) * LDS_STRIDE + squad * 8] = p1;
      }
    }
  }

  // epilogue: C layout — VGPR r: lanes 0-15 => M=r, lanes 16-31 => M=8+r; N = lane&15
  const int rbase = blockRow + m0 + ((lane >> 4) ? 8 : 0);
  const int cn = lane & 15;
#pragma unroll
  for (int mf = 0; mf < 2; ++mf) {
#pragma unroll
    for (int nf = 0; nf < 4; ++nf) {
      int col = (nt0 + nf) * 16 + cn;
      float bv = bias ? bias[col] : 0.0f;
#pragma unroll
      for (int r = 0; r < 8; ++r) {
        float c = acc[mf][nf][r] + bv;
        if (relu) c = fmaxf(c, 0.0f);
        out[(size_t)(rbase + mf * 16 + r) * HDIM + col] = f2bf(c);
      }
    }
  }
}

// =======================================================================
extern "C" void kernel_launch(void* const* d_in, const int* in_sizes, int n_in,
                              void* d_out, int out_size, void* d_ws, size_t ws_size,
                              hipStream_t stream) {
  const float* pos_in = (const float*)d_in[0];
  const float* node_f = (const float*)d_in[1];
  const float* mass   = (const float*)d_in[2];
  const float* nW1 = (const float*)d_in[3];  const float* nb1 = (const float*)d_in[4];
  const float* nW2 = (const float*)d_in[5];  const float* nb2 = (const float*)d_in[6];
  const float* eW1 = (const float*)d_in[7];  const float* eb1 = (const float*)d_in[8];
  const float* eW2 = (const float*)d_in[9];  const float* eb2 = (const float*)d_in[10];
  const float* rmW1 = (const float*)d_in[11]; const float* rmb1 = (const float*)d_in[12];
  const float* rmW2 = (const float*)d_in[13]; const float* rmb2 = (const float*)d_in[14];
  const float* rmW3 = (const float*)d_in[15]; const float* rmb3 = (const float*)d_in[16];
  const float* rmW4 = (const float*)d_in[17]; const float* rmb4 = (const float*)d_in[18];
  const float* omW1 = (const float*)d_in[19]; const float* omb1 = (const float*)d_in[20];
  const float* omW2 = (const float*)d_in[21]; const float* omb2 = (const float*)d_in[22];
  const float* dW1  = (const float*)d_in[23]; const float* db1  = (const float*)d_in[24];
  const float* dW2  = (const float*)d_in[25]; const float* db2  = (const float*)d_in[26];

  // ---- workspace carve-out ----
  char* cur = (char*)d_ws;
  auto walloc = [&](size_t bytes) -> void* {
    void* p = (void*)cur;
    cur += (bytes + 255) & ~(size_t)255;
    return p;
  };
  float* pos  = (float*)walloc(N_ATOMS * 3 * sizeof(float));
  float* vels = (float*)walloc(N_ATOMS * 3 * sizeof(float));
  int* senders = (int*)walloc(E_EDGES * sizeof(int));
  unsigned short* ea0  = (unsigned short*)walloc((size_t)E_EDGES * 4 * 2);
  unsigned short* xA   = (unsigned short*)walloc((size_t)N_ATOMS * HDIM * 2);
  unsigned short* xB   = (unsigned short*)walloc((size_t)N_ATOMS * HDIM * 2);
  unsigned short* xh   = (unsigned short*)walloc((size_t)N_ATOMS * HDIM * 2);
  unsigned short* xdec = (unsigned short*)walloc((size_t)N_ATOMS * HDIM * 2);
  unsigned short* ea   = (unsigned short*)walloc((size_t)E_EDGES * HDIM * 2);
  unsigned short* h1   = (unsigned short*)walloc((size_t)E_EDGES * HDIM * 2);
  unsigned short* h2   = (unsigned short*)walloc((size_t)E_EDGES * HDIM * 2);
  float* agg = (float*)walloc((size_t)N_ATOMS * HDIM * sizeof(float));

  // packed weights (dwords): per 32-K-block = 8*32*8 = 2048 dwords
  auto palloc = [&](size_t dwords) -> unsigned* { return (unsigned*)walloc(dwords * 4); };
  unsigned* nW1p = palloc(2048);            // K=24 -> pad 32
  unsigned* nW2p = palloc(4 * 2048);
  unsigned* eW1p = palloc(2048);            // K=4  -> pad 32
  unsigned* eW2p = palloc(4 * 2048);
  unsigned* rm1p = palloc((size_t)L_LAYERS * 12 * 2048);  // K=384
  unsigned* rm2p = palloc((size_t)L_LAYERS * 4 * 2048);
  unsigned* rm3p = palloc((size_t)L_LAYERS * 4 * 2048);
  unsigned* rm4p = palloc((size_t)L_LAYERS * 4 * 2048);
  unsigned* om1p = palloc((size_t)L_LAYERS * 8 * 2048);   // K=256
  unsigned* om2p = palloc((size_t)L_LAYERS * 4 * 2048);
  unsigned* dW1p = palloc(4 * 2048);

  auto pack = [&](const float* W, unsigned* dst, int Kvalid) {
    int kblocks = (Kvalid + 31) / 32;
    pack_w_kernel<<<dim3(kblocks), dim3(256), 0, stream>>>(W, dst, Kvalid);
  };
  pack(nW1, nW1p, 24);
  pack(nW2, nW2p, 128);
  pack(eW1, eW1p, 4);
  pack(eW2, eW2p, 128);
  for (int l = 0; l < L_LAYERS; ++l) {
    pack(rmW1 + (size_t)l * 384 * HDIM, rm1p + (size_t)l * 12 * 2048, 384);
    pack(rmW2 + (size_t)l * 128 * HDIM, rm2p + (size_t)l * 4 * 2048, 128);
    pack(rmW3 + (size_t)l * 128 * HDIM, rm3p + (size_t)l * 4 * 2048, 128);
    pack(rmW4 + (size_t)l * 128 * HDIM, rm4p + (size_t)l * 4 * 2048, 128);
    pack(omW1 + (size_t)l * 256 * HDIM, om1p + (size_t)l * 8 * 2048, 256);
    pack(omW2 + (size_t)l * 128 * HDIM, om2p + (size_t)l * 4 * 2048, 128);
  }

  GemmSeg znull = {nullptr, nullptr, nullptr, 0, 0, 0, 0};
  auto gemm1 = [&](const void* A, int K, int lda, int dtype, const unsigned* Wp,
                   const float* bias, int relu, unsigned short* out, int M,
                   const int* idx, int mode) {
    GemmSeg sg = {A, idx, Wp, K, lda, dtype, mode};
    gemm128_kernel<<<dim3(M / 128), dim3(256), 0, stream>>>(sg, znull, znull, 1, bias, relu, out);
  };

  init_kernel<<<dim3((N_ATOMS * 3 + 255) / 256), dim3(256), 0, stream>>>(pos_in, pos, vels);

  unsigned short* xc = xA;
  unsigned short* xn = xB;
  for (int step = 0; step < N_STEPS; ++step) {
    knn_kernel<<<dim3(N_ATOMS / 256), dim3(256), 0, stream>>>(pos, senders);
    edge_attr_kernel<<<dim3(E_EDGES / 256), dim3(256), 0, stream>>>(pos, senders, ea0);

    // node encoder: x = relu(node_f@nW1+nb1)@nW2+nb2
    gemm1(node_f, 24, 24, 1, nW1p, nb1, 1, xh, N_ATOMS, nullptr, 0);
    gemm1(xh, 128, 128, 0, nW2p, nb2, 0, xc, N_ATOMS, nullptr, 0);
    // edge encoder: ea = relu(ea0@eW1+eb1)@eW2+eb2
    gemm1(ea0, 4, 4, 0, eW1p, eb1, 1, h1, E_EDGES, nullptr, 0);
    gemm1(h1, 128, 128, 0, eW2p, eb2, 0, ea, E_EDGES, nullptr, 0);

    for (int l = 0; l < L_LAYERS; ++l) {
      const unsigned* w1 = rm1p + (size_t)l * 12 * 2048;
      // relation layer 1: fused concat(x[senders], x[receivers], ea) @ W1
      {
        GemmSeg sa = {xc, senders, w1,            128, 128, 0, 1};  // x[senders]
        GemmSeg sb = {xc, nullptr, w1 + 4 * 2048, 128, 128, 0, 2};  // x[row>>4] receivers
        GemmSeg sc = {ea, nullptr, w1 + 8 * 2048, 128, 128, 0, 0};  // ea
        gemm128_kernel<<<dim3(E_EDGES / 128), dim3(256), 0, stream>>>(
            sa, sb, sc, 3, rmb1 + l * HDIM, 1, h1);
      }
      gemm1(h1, 128, 128, 0, rm2p + (size_t)l * 4 * 2048, rmb2 + l * HDIM, 1, h2, E_EDGES, nullptr, 0);
      gemm1(h2, 128, 128, 0, rm3p + (size_t)l * 4 * 2048, rmb3 + l * HDIM, 1, h1, E_EDGES, nullptr, 0);
      gemm1(h1, 128, 128, 0, rm4p + (size_t)l * 4 * 2048, rmb4 + l * HDIM, 1, ea, E_EDGES, nullptr, 0);

      // agg = segment_sum(e, senders)
      hipMemsetAsync(agg, 0, (size_t)N_ATOMS * HDIM * sizeof(float), stream);
      scatter_kernel<<<dim3(E_EDGES / 256), dim3(256), 0, stream>>>(ea, senders, agg);

      // object model: x_new = relu(concat(x, agg)@oW1+ob1)@oW2+ob2
      {
        const unsigned* o1 = om1p + (size_t)l * 8 * 2048;
        GemmSeg sa = {xc,  nullptr, o1,            128, 128, 0, 0};
        GemmSeg sb = {agg, nullptr, o1 + 4 * 2048, 128, 128, 1, 0};  // fp32 segment
        gemm128_kernel<<<dim3(N_ATOMS / 128), dim3(256), 0, stream>>>(
            sa, sb, znull, 2, omb1 + l * HDIM, 1, xh);
      }
      gemm1(xh, 128, 128, 0, om2p + (size_t)l * 4 * 2048, omb2 + l * HDIM, 0, xn, N_ATOMS, nullptr, 0);
      unsigned short* tmp = xc; xc = xn; xn = tmp;
    }

    // decoder + integrate
    gemm1(xc, 128, 128, 0, dW1p, db1, 1, xdec, N_ATOMS, nullptr, 0);
    decode_integrate_kernel<<<dim3(N_ATOMS / 256), dim3(256), 0, stream>>>(
        xdec, dW2, db2, mass, pos, vels);
  }

  hipMemcpyAsync(d_out, pos, (size_t)N_ATOMS * 3 * sizeof(float),
                 hipMemcpyDeviceToDevice, stream);
}
